// GRU_75780402970951
// MI455X (gfx1250) — compile-verified, matrix operand order checked by reading, not run on hardware
//
#include <hip/hip_runtime.h>
#include <hip/hip_bf16.h>
#include <math.h>

// ---------------------------------------------------------------------------
// 2-layer GRU (B=32, T=512, D_IN=512, H=1024) + FC (D_OUT=512) for gfx1250.
// bf16 WMMA (v_wmma_f32_16x16x32_bf16), f32 accumulation.
//  - x-projections: dense WMMA GEMM, 2x2 tiles/wave (4 acc, 4 frag loads/K-step)
//  - recurrence: persistent kernel, 32 resident WGs, W_hh slice resident in
//    LDS (loaded once via async-to-LDS when available), global atomic barrier
//    per timestep.
// ---------------------------------------------------------------------------

typedef __attribute__((ext_vector_type(16))) __bf16 v16bf;
typedef __attribute__((ext_vector_type(8)))  float  v8f;
typedef int v4i __attribute__((vector_size(16)));   // matches async-LDS builtin param

#define WMMA_BF16(a, b, c) \
  __builtin_amdgcn_wmma_f32_16x16x32_bf16(false, (a), false, (b), (short)0, (c), false, false)

// gfx1250 async load-to-LDS path (guarded: falls back to plain copies)
#if defined(__has_builtin)
#if __has_builtin(__builtin_amdgcn_global_load_async_to_lds_b128) && \
    __has_builtin(__builtin_amdgcn_s_wait_asynccnt)
#define HAVE_ASYNC_LDS 1
#endif
#endif
#ifndef HAVE_ASYNC_LDS
#define HAVE_ASYNC_LDS 0
#endif

static constexpr int kB    = 32;
static constexpr int kT    = 512;
static constexpr int kDIN  = 512;
static constexpr int kH    = 1024;
static constexpr int kDOUT = 512;
static constexpr int k3H   = 3 * kH;

// recurrence tiling
static constexpr int RGRID   = 32;            // persistent workgroups
static constexpr int RCOLS   = kH / RGRID;    // 32 h'-columns per WG
static constexpr int RW_ROWS = 3 * RCOLS;     // 96 weight rows in LDS
static constexpr int RSTRIDE = kH + 8;        // padded LDS row (bank-conflict break)

// ---------------------------------------------------------------------------
// f32 -> bf16 flat convert (weights)
// ---------------------------------------------------------------------------
__global__ void cvt_f32_bf16(const float* __restrict__ src, __bf16* __restrict__ dst, size_t n) {
  size_t i = (size_t)blockIdx.x * blockDim.x + threadIdx.x;
  if (i < n) dst[i] = (__bf16)src[i];
}

// x [B,T,D] f32 -> x_t [T,B,D] bf16 (transpose for time-major GEMM)
__global__ void cvt_x_transpose(const float* __restrict__ x, __bf16* __restrict__ xt) {
  size_t i = (size_t)blockIdx.x * blockDim.x + threadIdx.x;
  size_t total = (size_t)kB * kT * kDIN;
  if (i >= total) return;
  int d = (int)(i % kDIN);
  size_t bt = i / kDIN;
  int t = (int)(bt % kT);
  int b = (int)(bt / kT);
  xt[((size_t)t * kB + b) * kDIN + d] = (__bf16)x[i];
}

// ---------------------------------------------------------------------------
// C_bf16[M,N] = A_bf16[M,K] @ W_bf16[N,K]^T + bias_f32[N]
// Block tile 64x128, 256 threads (8 waves), wave grid 2(M)x4(N),
// each wave owns a 2x2 grid of 16x16 tiles: 4 WMMAs per 4 fragment loads.
// ---------------------------------------------------------------------------
__global__ __launch_bounds__(256) void gemm_bf16_bias(
    const __bf16* __restrict__ A, const __bf16* __restrict__ W,
    const float* __restrict__ bias, __bf16* __restrict__ C,
    int M, int N, int K) {
  const int lane = threadIdx.x & 31;
  const int wave = threadIdx.x >> 5;
  const int wm = wave & 1;        // M half (2 subtiles each)
  const int wn = wave >> 1;       // N quarter (2 subtiles each)
  const int bm = blockIdx.x * 64;
  const int bn = blockIdx.y * 128;
  const int r  = lane & 15;
  const int kh = lane >> 4;

  const __bf16* ap0 = A + (size_t)(bm + (wm * 2 + 0) * 16 + r) * K + kh * 16;
  const __bf16* ap1 = A + (size_t)(bm + (wm * 2 + 1) * 16 + r) * K + kh * 16;
  const __bf16* bp0 = W + (size_t)(bn + (wn * 2 + 0) * 16 + r) * K + kh * 16;
  const __bf16* bp1 = W + (size_t)(bn + (wn * 2 + 1) * 16 + r) * K + kh * 16;

  v8f acc00 = {}, acc01 = {}, acc10 = {}, acc11 = {};
  for (int k0 = 0; k0 < K; k0 += 32) {
    if (k0 + 128 < K) __builtin_prefetch(ap0 + k0 + 128, 0, 3);
    v16bf a0 = *(const v16bf*)(ap0 + k0);
    v16bf a1 = *(const v16bf*)(ap1 + k0);
    v16bf b0 = *(const v16bf*)(bp0 + k0);
    v16bf b1 = *(const v16bf*)(bp1 + k0);
    acc00 = WMMA_BF16(a0, b0, acc00);
    acc01 = WMMA_BF16(a0, b1, acc01);
    acc10 = WMMA_BF16(a1, b0, acc10);
    acc11 = WMMA_BF16(a1, b1, acc11);
  }

  const int n0 = bn + (wn * 2 + 0) * 16 + r;
  const int n1 = bn + (wn * 2 + 1) * 16 + r;
  const int m0 = bm + (wm * 2 + 0) * 16 + kh * 8;
  const int m1 = bm + (wm * 2 + 1) * 16 + kh * 8;
  const float bi0 = bias[n0], bi1 = bias[n1];
#pragma unroll
  for (int i = 0; i < 8; ++i) {
    C[(size_t)(m0 + i) * N + n0] = (__bf16)(acc00[i] + bi0);
    C[(size_t)(m0 + i) * N + n1] = (__bf16)(acc01[i] + bi1);
    C[(size_t)(m1 + i) * N + n0] = (__bf16)(acc10[i] + bi0);
    C[(size_t)(m1 + i) * N + n1] = (__bf16)(acc11[i] + bi1);
  }
}

// ---------------------------------------------------------------------------
// Persistent GRU recurrence for one layer.
//   xproj [T, B, 3H] bf16 (already includes b_ih)
//   Whh   [3H, H]    bf16 row-major
//   bhh   [3H]       f32
//   hseq  [T, B, H]  bf16 (row t-1 is the A-matrix for step t; h0 for t=0)
// Grid = RGRID WGs x 128 threads (4 waves). Each WG owns RCOLS=32 columns of
// h'; its W_hh slice (96 rows x 1024, padded) lives in LDS for all 512 steps.
// Each wave owns one 16x16 (m,n) tile and its 3 gate accumulators.
// ---------------------------------------------------------------------------
__global__ __launch_bounds__(128) void gru_recurrence(
    const __bf16* __restrict__ xproj, const __bf16* __restrict__ Whh,
    const float* __restrict__ bhh, const __bf16* __restrict__ h0,
    __bf16* __restrict__ hseq, unsigned* __restrict__ barrier_cnt, int T) {
  __shared__ __align__(32) __bf16 w_sh[RW_ROWS * RSTRIDE];  // ~193.5 KB

  const int tid  = threadIdx.x;
  const int lane = tid & 31;
  const int wave = tid >> 5;
  const int mt = wave & 1;                 // batch rows 0-15 / 16-31
  const int nt = wave >> 1;                // 0..1 -> 16-col subtile
  const int colbase = blockIdx.x * RCOLS;
  const int r  = lane & 15;
  const int kh = lane >> 4;

  // ---- one-time: stage W_hh slice into LDS --------------------------------
  // LDS row lr = g*RCOLS + lc  <->  global row g*kH + colbase + lc.
  // 128 uint4 of data per row; LDS row stride = 129 uint4 (2064 B, padded).
  {
    const int nchunks = RW_ROWS * 128;
    for (int i = tid; i < nchunks; i += 128) {
      int lr = i >> 7;
      int ci = i & 127;
      int g  = lr / RCOLS;
      int lc = lr - g * RCOLS;
      const uint4* src = (const uint4*)(Whh + (size_t)(g * kH + colbase + lc) * kH) + ci;
      uint4* dst = (uint4*)w_sh + (size_t)lr * 129 + ci;
#if HAVE_ASYNC_LDS
      __builtin_amdgcn_global_load_async_to_lds_b128((v4i*)src, (v4i*)dst, 0, 0);
#else
      *dst = *src;
#endif
    }
#if HAVE_ASYNC_LDS
    __builtin_amdgcn_s_wait_asynccnt(0);
#endif
  }
  __syncthreads();

  const __bf16* wr = w_sh + (size_t)(0 * RCOLS + nt * 16 + r) * RSTRIDE + kh * 16;
  const __bf16* wz = w_sh + (size_t)(1 * RCOLS + nt * 16 + r) * RSTRIDE + kh * 16;
  const __bf16* wn = w_sh + (size_t)(2 * RCOLS + nt * 16 + r) * RSTRIDE + kh * 16;

  const int n = colbase + nt * 16 + r;     // output column within H
  const float bhr = bhh[0 * kH + n];
  const float bhz = bhh[1 * kH + n];
  const float bhn = bhh[2 * kH + n];

  for (int t = 0; t < T; ++t) {
    const __bf16* hprev = (t == 0) ? h0 : (hseq + (size_t)(t - 1) * kB * kH);
    const __bf16* ap = hprev + (size_t)(mt * 16 + r) * kH + kh * 16;

    // ---- hp = h @ Whh^T for 3 gate tiles (A from L2-hot global, B from LDS)
    v8f accR = {}, accZ = {}, accN = {};
    for (int k0 = 0; k0 < kH; k0 += 32) {
      v16bf a  = *(const v16bf*)(ap + k0);
      v16bf br = *(const v16bf*)(wr + k0);
      v16bf bz = *(const v16bf*)(wz + k0);
      v16bf bn = *(const v16bf*)(wn + k0);
      accR = WMMA_BF16(a, br, accR);
      accZ = WMMA_BF16(a, bz, accZ);
      accN = WMMA_BF16(a, bn, accN);
    }

    // ---- gating epilogue (PyTorch GRUCell semantics) ------------------
    const __bf16* xp = xproj + (size_t)t * kB * k3H;
    __bf16* hout = hseq + (size_t)t * kB * kH;
#pragma unroll
    for (int i = 0; i < 8; ++i) {
      int m = mt * 16 + kh * 8 + i;        // batch row
      float xr = (float)xp[(size_t)m * k3H + 0 * kH + n];
      float xz = (float)xp[(size_t)m * k3H + 1 * kH + n];
      float xn = (float)xp[(size_t)m * k3H + 2 * kH + n];
      float hr = accR[i] + bhr;
      float hz = accZ[i] + bhz;
      float hn = accN[i] + bhn;            // b_hh_n stays inside r*(...)
      float rg = 1.0f / (1.0f + __expf(-(xr + hr)));
      float zg = 1.0f / (1.0f + __expf(-(xz + hz)));
      float ng = tanhf(xn + rg * hn);
      float hold = (float)hprev[(size_t)m * kH + n];
      hout[(size_t)m * kH + n] = (__bf16)((1.0f - zg) * ng + zg * hold);
    }

    // ---- device-wide barrier before next step -------------------------
    __threadfence();                       // flush this thread's h stores
    __syncthreads();
    if (tid == 0) {
      atomicAdd(barrier_cnt, 1u);
      unsigned target = (unsigned)(t + 1) * gridDim.x;
      while (__hip_atomic_load(barrier_cnt, __ATOMIC_ACQUIRE,
                               __HIP_MEMORY_SCOPE_AGENT) < target)
        __builtin_amdgcn_s_sleep(1);
    }
    __syncthreads();
  }
}

// ---------------------------------------------------------------------------
// out_f32[32,512] = h_last_bf16[32,1024] @ Wfc_bf16[512,1024]^T + b_fc
// Grid = 8 WGs of 256 threads; one 16x16 tile per wave.
// ---------------------------------------------------------------------------
__global__ __launch_bounds__(256) void fc_out(
    const __bf16* __restrict__ A, const __bf16* __restrict__ W,
    const float* __restrict__ bias, float* __restrict__ out) {
  const int lane = threadIdx.x & 31;
  const int wave = threadIdx.x >> 5;
  const int mt = wave & 1;
  const int nt = wave >> 1;
  const int bn = blockIdx.x * 64;
  const int r  = lane & 15;
  const int kh = lane >> 4;

  const __bf16* ap = A + (size_t)(mt * 16 + r) * kH + kh * 16;
  const __bf16* bp = W + (size_t)(bn + nt * 16 + r) * kH + kh * 16;

  v8f acc = {};
  for (int k0 = 0; k0 < kH; k0 += 32) {
    v16bf a = *(const v16bf*)(ap + k0);
    v16bf b = *(const v16bf*)(bp + k0);
    acc = WMMA_BF16(a, b, acc);
  }

  const int n = bn + nt * 16 + r;
  const float bi = bias[n];
#pragma unroll
  for (int i = 0; i < 8; ++i) {
    int m = mt * 16 + kh * 8 + i;
    out[(size_t)m * kDOUT + n] = acc[i] + bi;
  }
}

// ---------------------------------------------------------------------------
// Host side
// ---------------------------------------------------------------------------
extern "C" void kernel_launch(void* const* d_in, const int* in_sizes, int n_in,
                              void* d_out, int out_size, void* d_ws, size_t ws_size,
                              hipStream_t stream) {
  const float* x     = (const float*)d_in[0];
  const float* Wih0  = (const float*)d_in[1];
  const float* Whh0  = (const float*)d_in[2];
  const float* bih0  = (const float*)d_in[3];
  const float* bhh0  = (const float*)d_in[4];
  const float* Wih1  = (const float*)d_in[5];
  const float* Whh1  = (const float*)d_in[6];
  const float* bih1  = (const float*)d_in[7];
  const float* bhh1  = (const float*)d_in[8];
  const float* Wfc   = (const float*)d_in[9];
  const float* bfc   = (const float*)d_in[10];
  float* out = (float*)d_out;

  // workspace layout (256B aligned)
  char* ws = (char*)d_ws;
  size_t off = 0;
  auto alloc = [&](size_t bytes) -> char* {
    char* p = ws + off;
    off += (bytes + 255) & ~(size_t)255;
    return p;
  };
  __bf16* xt_bf    = (__bf16*)alloc((size_t)kT * kB * kDIN * 2);   // 16.8 MB
  __bf16* wih0_bf  = (__bf16*)alloc((size_t)k3H * kDIN * 2);       //  3.1 MB
  __bf16* whh0_bf  = (__bf16*)alloc((size_t)k3H * kH * 2);         //  6.3 MB
  __bf16* wih1_bf  = (__bf16*)alloc((size_t)k3H * kH * 2);
  __bf16* whh1_bf  = (__bf16*)alloc((size_t)k3H * kH * 2);
  __bf16* wfc_bf   = (__bf16*)alloc((size_t)kDOUT * kH * 2);       //  1.0 MB
  __bf16* xproj    = (__bf16*)alloc((size_t)kT * kB * k3H * 2);    // 100.7 MB (reused both layers)
  __bf16* h1seq    = (__bf16*)alloc((size_t)kT * kB * kH * 2);     // 33.6 MB
  __bf16* h2seq    = (__bf16*)alloc((size_t)kT * kB * kH * 2);     // 33.6 MB
  __bf16* h0zero   = (__bf16*)alloc((size_t)kB * kH * 2);          // 64 KB
  unsigned* cnt    = (unsigned*)alloc(256);

  auto cvt = [&](const float* s, __bf16* d, size_t n) {
    cvt_f32_bf16<<<(unsigned)((n + 255) / 256), 256, 0, stream>>>(s, d, n);
  };

  // --- precision conversion ------------------------------------------------
  {
    size_t n = (size_t)kB * kT * kDIN;
    cvt_x_transpose<<<(unsigned)((n + 255) / 256), 256, 0, stream>>>(x, xt_bf);
  }
  cvt(Wih0, wih0_bf, (size_t)k3H * kDIN);
  cvt(Whh0, whh0_bf, (size_t)k3H * kH);
  cvt(Wih1, wih1_bf, (size_t)k3H * kH);
  cvt(Whh1, whh1_bf, (size_t)k3H * kH);
  cvt(Wfc,  wfc_bf,  (size_t)kDOUT * kH);
  (void)hipMemsetAsync(h0zero, 0, (size_t)kB * kH * 2, stream);

  const int Mrows = kT * kB;  // 16384

  // --- layer 0: x-projection + recurrence ----------------------------------
  gemm_bf16_bias<<<dim3(Mrows / 64, k3H / 128), 256, 0, stream>>>(
      xt_bf, wih0_bf, bih0, xproj, Mrows, k3H, kDIN);
  (void)hipMemsetAsync(cnt, 0, 256, stream);
  gru_recurrence<<<RGRID, 128, 0, stream>>>(xproj, whh0_bf, bhh0, h0zero, h1seq, cnt, kT);

  // --- layer 1: x-projection + recurrence ----------------------------------
  gemm_bf16_bias<<<dim3(Mrows / 64, k3H / 128), 256, 0, stream>>>(
      h1seq, wih1_bf, bih1, xproj, Mrows, k3H, kH);
  (void)hipMemsetAsync(cnt, 0, 256, stream);
  gru_recurrence<<<RGRID, 128, 0, stream>>>(xproj, whh1_bf, bhh1, h0zero, h2seq, cnt, kT);

  // --- final FC on last hidden state ---------------------------------------
  const __bf16* hlast = h2seq + (size_t)(kT - 1) * kB * kH;
  fc_out<<<kDOUT / 64, 256, 0, stream>>>(hlast, wfc_bf, bfc, out);
}
